// DCN_74775380623842
// MI455X (gfx1250) — matrix operand order repaired and checked by reading, not compiled, not run on hardware
//
#include <hip/hip_runtime.h>
#include <hip/hip_bf16.h>

typedef __attribute__((ext_vector_type(16))) _Float16 v16h;
typedef __attribute__((ext_vector_type(8)))  _Float16 v8h;
typedef __attribute__((ext_vector_type(8)))  float    v8f;

#define B_      8
#define CIN     64
#define COUT    64
#define H_      128
#define W_      128
#define HW      (H_ * W_)
#define K9      9
#define KDIM    576            // 9 * 64 contraction length
#define KSTEPS  18             // 576 / 32
#define TILE_N  32             // pixels per workgroup
#define NTILES  2              // 32 / 16
#define PCH     32             // padded param channels (27 -> 32)

// ---- A-matrix (16-bit, 16x32) fragment element -> K mapping (contiguous) ----
// lanes 0-15: M=0..15, element e: K = e
// lanes 16-31: same M, K = 16 + e (matches B fragment layout).
__device__ __host__ __forceinline__ int a_frag_kk(int lane, int e) {
    return e + ((lane >> 4) << 4);
}

// ---------------- kernel 1: x NCHW fp32 -> NHWC f16 (16B stores) ----------------
__global__ void dcn_nhwc_f16(const float* __restrict__ x, _Float16* __restrict__ xh) {
    int p = blockIdx.x * blockDim.x + threadIdx.x;     // global pixel id
    if (p >= B_ * HW) return;
    int b = p / HW, pp = p - b * HW;
    const float* src = x + (size_t)b * CIN * HW + pp;  // lane-coalesced reads
    _Float16* dst = xh + (size_t)p * CIN;
    for (int c0 = 0; c0 < CIN; c0 += 8) {
        v8h v;
#pragma unroll
        for (int j = 0; j < 8; ++j) v[j] = (_Float16)src[(c0 + j) * HW];
        *(v8h*)(dst + c0) = v;                         // 16B store
    }
}

// ---------------- kernel 2: pack weights into A-fragment layout ----------------
__global__ void dcn_pack_w(const float* __restrict__ weight,
                           const float* __restrict__ pg_weight,
                           _Float16* __restrict__ Amain,
                           _Float16* __restrict__ Apg) {
    const int NMAIN = 4 * KSTEPS * 32 * 16;   // 64 rows -> 4 M-tiles
    const int NPG   = 2 * KSTEPS * 32 * 16;   // 27 rows -> 2 M-tiles (padded)
    int idx = blockIdx.x * blockDim.x + threadIdx.x;
    if (idx >= NMAIN + NPG) return;
    bool is_pg = idx >= NMAIN;
    int li   = is_pg ? idx - NMAIN : idx;     // ((mt*KSTEPS+ks)*32+lane)*16+e
    int e    = li & 15;
    int lane = (li >> 4) & 31;
    int ks   = (li >> 9) % KSTEPS;
    int mt   = (li >> 9) / KSTEPS;
    int m    = mt * 16 + (lane & 15);
    int K    = ks * 32 + a_frag_kk(lane, e);
    int k    = K >> 6, c = K & 63;
    int ki   = k / 3, kj = k % 3;             // OIHW flat: o*576 + c*9 + ki*3 + kj
    if (is_pg) {
        float v = (m < 27) ? pg_weight[m * 576 + c * 9 + ki * 3 + kj] : 0.f;
        Apg[li] = (_Float16)v;
    } else {
        Amain[li] = (_Float16)weight[m * 576 + c * 9 + ki * 3 + kj];
    }
}

// ---------------- kernel 3: offset/mask conv via WMMA ----------------
__global__ __launch_bounds__(256) void dcn_params(
    const _Float16* __restrict__ xh, const _Float16* __restrict__ Apg,
    const float* __restrict__ pg_bias, float* __restrict__ params) {
    __shared__ _Float16 sB[NTILES * KSTEPS * 32 * 16];   // B fragments, 36 KB
    int tid = threadIdx.x;
    int b    = blockIdx.x >> 9;          // 512 tiles per image
    int tile = blockIdx.x & 511;
    int pixbase = tile * TILE_N;
    int y     = pixbase >> 7;            // tile stays inside one row (32 | 128)
    int xbase = pixbase & 127;

    // dense im2col fill, 8 channels per thread-iteration (16B load + 16B ds store)
    for (int idx = tid; idx < TILE_N * K9 * 8; idx += 256) {
        int n  = idx / 72;               // pixel in tile
        int rr = idx - n * 72;
        int k  = rr >> 3;                // tap 0..8
        int c  = (rr & 7) << 3;          // channel group base
        int yy = y + k / 3 - 1, xx = xbase + n + (k % 3) - 1;
        v8h val = {};
        if (yy >= 0 && yy < H_ && xx >= 0 && xx < W_)
            val = *(const v8h*)(xh + (((b * H_ + yy) * W_ + xx) * CIN + c));
        int r = k * 64 + c;              // K index; group stays in one 32-block
        int kstep = r >> 5, kk = r & 31;
        int nt = n >> 4, nc = n & 15;
        int bl = nc + ((kk >> 4) << 4);  // B: lanes 0-15 K<16, 16-31 K>=16
        *(v8h*)(sB + ((nt * KSTEPS + kstep) * 32 + bl) * 16 + (kk & 15)) = val;
    }
    __syncthreads();

    int wv = tid >> 5, lane = tid & 31;
    if (wv < 4) {                                // 2 M-tiles x 2 N-tiles
        int mt = wv >> 1, nt = wv & 1;
        v8f acc = {};
        for (int ks = 0; ks < KSTEPS; ++ks) {
            v16h a  = *(const v16h*)(Apg + ((mt * KSTEPS + ks) * 32 + lane) * 16);
            v16h bb = *(const v16h*)(sB  + ((nt * KSTEPS + ks) * 32 + lane) * 16);
            acc = __builtin_amdgcn_wmma_f32_16x16x32_f16(false, a, false, bb,
                                                         (short)0, acc, false, false);
        }
        int pix = pixbase + nt * 16 + (lane & 15);
        int hi8 = (lane >= 16) ? 8 : 0;          // C/D: VGPR r -> M = r (+8 hi lanes)
        for (int r = 0; r < 8; ++r) {
            int ch = mt * 16 + r + hi8;
            if (ch < 27) {
                float v = acc[r] + pg_bias[ch];
                if (ch >= 18) v = 1.f / (1.f + __expf(-v));   // mask sigmoid
                params[(b * HW + pix) * PCH + ch] = v;
            }
        }
    }
}

// ---------------- kernel 4: deformable sampling + WMMA contraction ----------------
__global__ __launch_bounds__(256) void dcn_main(
    const _Float16* __restrict__ xh, const float* __restrict__ params,
    const _Float16* __restrict__ Amain, const float* __restrict__ bias,
    float* __restrict__ out) {
    __shared__ _Float16 sB[NTILES * KSTEPS * 32 * 16];   // 36 KB sampled matrix
    __shared__ int   sIdx[TILE_N * K9][4];               // clamped iy0,iy1,ix0,ix1
    __shared__ float sWt [TILE_N * K9][4];               // bilinear*valid*mask
    int tid = threadIdx.x;
    int b    = blockIdx.x >> 9;
    int tile = blockIdx.x & 511;
    int pixbase = tile * TILE_N;
    int y     = pixbase >> 7;
    int xbase = pixbase & 127;
    __builtin_prefetch(Amain, 0, 0);

    // per-(pixel, tap) bilinear setup; ref layout: dy=ch 2k, dx=ch 2k+1, mask=ch 18+k
    for (int q = tid; q < TILE_N * K9; q += 256) {
        int n = q / K9, k = q - n * K9;
        const float* pp = params + (b * HW + pixbase + n) * PCH;
        float dy = pp[2 * k], dx = pp[2 * k + 1], mk = pp[18 + k];
        float sy = (float)(y + k / 3 - 1) + dy;
        float sx = (float)(xbase + n + (k % 3) - 1) + dx;
        float fy = floorf(sy), fx = floorf(sx);
        int y0 = (int)fy, x0 = (int)fx;
        float ly = sy - fy, lx = sx - fx;
        float vy0 = (y0 >= 0 && y0 < H_) ? 1.f : 0.f;
        float vy1 = (y0 + 1 >= 0 && y0 + 1 < H_) ? 1.f : 0.f;
        float vx0 = (x0 >= 0 && x0 < W_) ? 1.f : 0.f;
        float vx1 = (x0 + 1 >= 0 && x0 + 1 < W_) ? 1.f : 0.f;
        sWt[q][0] = (1.f - ly) * (1.f - lx) * vy0 * vx0 * mk;
        sWt[q][1] = (1.f - ly) * lx         * vy0 * vx1 * mk;
        sWt[q][2] = ly * (1.f - lx)         * vy1 * vx0 * mk;
        sWt[q][3] = ly * lx                 * vy1 * vx1 * mk;
        sIdx[q][0] = min(max(y0, 0), H_ - 1);
        sIdx[q][1] = min(max(y0 + 1, 0), H_ - 1);
        sIdx[q][2] = min(max(x0, 0), W_ - 1);
        sIdx[q][3] = min(max(x0 + 1, 0), W_ - 1);
    }
    __syncthreads();

    // deformable im2col: 8 channels per iteration -> 4 x 16B gathers, fp32 blend,
    // one 16B LDS store straight into B-fragment layout
    for (int idx = tid; idx < TILE_N * K9 * 8; idx += 256) {
        int n  = idx / 72;
        int rr = idx - n * 72;
        int k  = rr >> 3;
        int c  = (rr & 7) << 3;
        int q  = n * K9 + k;
        int iy0 = sIdx[q][0], iy1 = sIdx[q][1], ix0 = sIdx[q][2], ix1 = sIdx[q][3];
        float w0 = sWt[q][0], w1 = sWt[q][1], w2 = sWt[q][2], w3 = sWt[q][3];
        const _Float16* base = xh + (b * HW) * CIN + c;
        v8h t00 = *(const v8h*)(base + (iy0 * W_ + ix0) * CIN);
        v8h t01 = *(const v8h*)(base + (iy0 * W_ + ix1) * CIN);
        v8h t10 = *(const v8h*)(base + (iy1 * W_ + ix0) * CIN);
        v8h t11 = *(const v8h*)(base + (iy1 * W_ + ix1) * CIN);
        v8h val;
#pragma unroll
        for (int j = 0; j < 8; ++j)
            val[j] = (_Float16)(w0 * (float)t00[j] + w1 * (float)t01[j]
                              + w2 * (float)t10[j] + w3 * (float)t11[j]);
        int r = k * 64 + c;
        int kstep = r >> 5, kk = r & 31;
        int nt = n >> 4, nc = n & 15;
        int bl = nc + ((kk >> 4) << 4);
        *(v8h*)(sB + ((nt * KSTEPS + kstep) * 32 + bl) * 16 + (kk & 15)) = val;
    }
    __syncthreads();

    // 8 waves = 4 M-tiles x 2 N-tiles, 18 WMMAs each over K=576
    int wv = tid >> 5, lane = tid & 31;
    int mt = wv >> 1, nt = wv & 1;
    v8f acc = {};
    for (int ks = 0; ks < KSTEPS; ++ks) {
        v16h a  = *(const v16h*)(Amain + ((mt * KSTEPS + ks) * 32 + lane) * 16);
        v16h bb = *(const v16h*)(sB    + ((nt * KSTEPS + ks) * 32 + lane) * 16);
        acc = __builtin_amdgcn_wmma_f32_16x16x32_f16(false, a, false, bb,
                                                     (short)0, acc, false, false);
    }
    int pix = pixbase + nt * 16 + (lane & 15);
    int hi8 = (lane >= 16) ? 8 : 0;
    for (int r = 0; r < 8; ++r) {
        int ch = mt * 16 + r + hi8;
        out[(b * COUT + ch) * HW + pix] = acc[r] + bias[ch];
    }
}

extern "C" void kernel_launch(void* const* d_in, const int* in_sizes, int n_in,
                              void* d_out, int out_size, void* d_ws, size_t ws_size,
                              hipStream_t stream) {
    const float* x         = (const float*)d_in[0];   // [8,64,128,128]
    const float* weight    = (const float*)d_in[1];   // [64,64,3,3]
    const float* bias      = (const float*)d_in[2];   // [64]
    const float* pg_weight = (const float*)d_in[3];   // [27,64,3,3]
    const float* pg_bias   = (const float*)d_in[4];   // [27]
    float* out = (float*)d_out;

    char* ws = (char*)d_ws;
    const size_t XH_BYTES    = (size_t)B_ * HW * CIN * 2;          // 16 MB f16 NHWC x
    const size_t AMAIN_BYTES = (size_t)4 * KSTEPS * 32 * 16 * 2;   // 72 KB
    const size_t APG_BYTES   = (size_t)2 * KSTEPS * 32 * 16 * 2;   // 36 KB
    _Float16* xh    = (_Float16*)(ws);
    _Float16* Amain = (_Float16*)(ws + XH_BYTES);
    _Float16* Apg   = (_Float16*)(ws + XH_BYTES + AMAIN_BYTES);
    float*    prm   = (float*)  (ws + XH_BYTES + AMAIN_BYTES + APG_BYTES); // 16 MB

    dcn_nhwc_f16<<<(B_ * HW + 255) / 256, 256, 0, stream>>>(x, xh);
    dcn_pack_w<<<(6 * KSTEPS * 32 * 16 + 255) / 256, 256, 0, stream>>>(
        weight, pg_weight, Amain, Apg);
    dcn_params<<<B_ * 512, 256, 0, stream>>>(xh, Apg, pg_bias, prm);
    dcn_main<<<B_ * 512, 256, 0, stream>>>(xh, prm, Amain, bias, out);
}
